// TriAttEnd_20194936225861
// MI455X (gfx1250) — matrix operand order repaired and checked by reading, not compiled, not run on hardware
//
#include <hip/hip_runtime.h>

typedef __bf16 v16bf __attribute__((ext_vector_type(16)));
typedef __bf16 v8bf  __attribute__((ext_vector_type(8)));
typedef float  v8f   __attribute__((ext_vector_type(8)));

#define LDIM 384
#define DDIM 128
#define HHEADS 12
#define CDIM 24
#define HC 288
#define LL (LDIM*LDIM)      // 147456
#define NQKVG 1152          // 4 * HC
#define CP 32               // C padded to 32 for clean K-chunks

// ---- workspace byte offsets (all 256-aligned) ----
#define OFF_WCAT 0ULL                                   // bf16 [1152][128]  (N-major)
#define OFF_WB16 (OFF_WCAT + 1152ULL*128*2)             // bf16 [16][128]    (N-major)
#define OFF_WO16 (OFF_WB16 + 16ULL*128*2)               // bf16 [128][288]   (N-major)
#define OFF_QP   (OFF_WO16 + 128ULL*288*2)              // bf16 [12][384][384][32]
#define OFF_KP   (OFF_QP + 12ULL*384*384*32*2)          // bf16 [12][384][384][32]
#define OFF_VA   (OFF_KP + 12ULL*384*384*32*2)          // bf16 [12][9216][384]  (j-major)
#define OFF_GB   (OFF_VA + 12ULL*9216*384*2)            // bf16 [147456][288]
#define OFF_PB   (OFF_GB + 147456ULL*288*2)             // f32  [147456][12]
#define OFF_ATT  (OFF_PB + 147456ULL*12*4)              // bf16 [12][384][384]
#define OFF_GO   (OFF_ATT + 12ULL*384*384*2)            // bf16 [147456][288]

__device__ __forceinline__ v16bf concat8(v8bf lo, v8bf hi) {
    return __builtin_shufflevector(lo, hi, 0,1,2,3,4,5,6,7,8,9,10,11,12,13,14,15);
}

// A-fragment for v_wmma_f32_16x16x32_bf16 from a contiguous 32-element K-chunk.
// Contiguous K order used here (B matches): lane half 0 -> K {0..15}, half 1 -> K {16..31}.
__device__ __forceinline__ v16bf load_a_frag(const __bf16* p, int half) {
    v8bf lo = *reinterpret_cast<const v8bf*>(p + half * 16);
    v8bf hi = *reinterpret_cast<const v8bf*>(p + half * 16 + 8);
    return concat8(lo, hi);
}

__device__ __forceinline__ v16bf load_b_frag(const __bf16* p) {
    return *reinterpret_cast<const v16bf*>(p);
}

// ---------------- weight conversion fp32 -> bf16 (transposed to N-major) ----------------
__global__ __launch_bounds__(256) void prep_weights(
    const float* Wq, const float* Wk, const float* Wv, const float* Wg,
    const float* Wb, const float* Wo,
    __bf16* WcatT, __bf16* Wb16T, __bf16* WoT)
{
    unsigned idx = blockIdx.x * 256 + threadIdx.x;
    const unsigned tot1 = NQKVG * 128;
    const unsigned tot2 = 16 * 128;
    const unsigned tot3 = 128 * HC;
    if (idx < tot1) {
        unsigned n = idx / 128, k = idx % 128;
        unsigned m = n / HC, nn = n % HC;
        const float* W = (m == 0) ? Wq : (m == 1) ? Wk : (m == 2) ? Wv : Wg;
        WcatT[idx] = (__bf16)W[k * HC + nn];
    } else if (idx < tot1 + tot2) {
        unsigned r = idx - tot1;
        unsigned n = r / 128, k = r % 128;
        Wb16T[r] = (__bf16)(n < HHEADS ? Wb[k * HHEADS + n] : 0.0f);
    } else if (idx < tot1 + tot2 + tot3) {
        unsigned r = idx - tot1 - tot2;
        unsigned n = r / HC, k = r % HC;
        WoT[r] = (__bf16)Wo[k * DDIM + n];
    }
}

// ---------------- zero the c=24..31 K-pads of q/k staging (one 16B store per row) ---------
__global__ __launch_bounds__(256) void zero_pads(__bf16* qp, __bf16* kp)
{
    unsigned idx = blockIdx.x * 256 + threadIdx.x;   // one per (h, l2, l1) row
    const unsigned rows = 12u * LDIM * LDIM;
    if (idx < rows) {
        v8bf zz = {};
        *reinterpret_cast<v8bf*>(qp + (size_t)idx * CP + CDIM) = zz;
        *reinterpret_cast<v8bf*>(kp + (size_t)idx * CP + CDIM) = zz;
    }
}

// ---------- fused LayerNorm + q/k/v/g/b projections ----------
// wave w handles tiles [(w&1)*9, +9) of matrix (w>>1): 18 tiles/matrix = 2 waves, balanced.
__global__ __launch_bounds__(256) void ln_qkvg(
    const float* z, const float* lng, const float* lnb,
    const float* bq, const float* bk, const float* bv, const float* bg,
    const float* bb2,
    const __bf16* WcatT, const __bf16* Wb16T,
    __bf16* qp, __bf16* kp, __bf16* va, __bf16* gb, float* projb)
{
    __shared__ __attribute__((aligned(16))) __bf16 As[16][136];
    int t = threadIdx.x;
    unsigned r0 = blockIdx.x * 16;
    unsigned l1u = r0 / LDIM;        // block-uniform (16-row tile never crosses l1)
    unsigned l2b = r0 % LDIM;        // row l2 = l2b + rowInTile
    {
        int i = t >> 4, j = t & 15;
        const float4* zr = reinterpret_cast<const float4*>(
            z + (unsigned)((l2b + i) * LDIM + l1u) * DDIM + j * 8);
        float4 a0 = zr[0], a1 = zr[1];
        float v[8] = {a0.x, a0.y, a0.z, a0.w, a1.x, a1.y, a1.z, a1.w};
        float s = 0.f, sq = 0.f;
#pragma unroll
        for (int e = 0; e < 8; ++e) { s += v[e]; sq += v[e] * v[e]; }
#pragma unroll
        for (int m = 8; m >= 1; m >>= 1) {
            s  += __shfl_xor(s,  m, 16);
            sq += __shfl_xor(sq, m, 16);
        }
        float mu = s * (1.0f / 128.0f);
        float var = sq * (1.0f / 128.0f) - mu * mu;
        float rstd = rsqrtf(var + 1e-5f);
        v8bf pack;
#pragma unroll
        for (int e = 0; e < 8; ++e) {
            int d = j * 8 + e;
            pack[e] = (__bf16)((v[e] - mu) * rstd * lng[d] + lnb[d]);
        }
        *reinterpret_cast<v8bf*>(&As[i][j * 8]) = pack;   // one ds_store_b128
    }
    __syncthreads();

    // readfirstlane -> SGPR -> scalar (non-divergent) branches on the matrix id
    int wave = __builtin_amdgcn_readfirstlane(t >> 5);
    int lane = t & 31, half = lane >> 4, ln = lane & 15;

    // Hoist the 4 A-fragments (shared across all N-tiles of this wave).
    v16bf afrag[4];
#pragma unroll
    for (int kc = 0; kc < 4; ++kc) afrag[kc] = load_a_frag(&As[ln][kc * 32], half);

    const int mmat = wave >> 1;          // matrix 0..3 (q,k,v,g) - scalar
    const unsigned tb = (wave & 1) * 9u; // first tile within matrix - scalar
    const __bf16* wbase = WcatT + ((unsigned)mmat * HC + tb * 16u + (unsigned)ln) * 128u
                        + half * 16;

    if (mmat <= 1) {
        __bf16* dst = (mmat == 0) ? qp : kp;
        const float* biasp = (mmat == 0) ? bq : bk;
        for (int it = 0; it < 9; ++it) {
            v8f acc = {};
            const __bf16* wcol = wbase + (unsigned)it * (16 * 128);
#pragma unroll
            for (int kc = 0; kc < 4; ++kc)
                acc = __builtin_amdgcn_wmma_f32_16x16x32_bf16(
                    false, afrag[kc], false, load_b_frag(wcol + kc * 32),
                    (short)0, acc, false, false);
            unsigned nn = (tb + (unsigned)it) * 16u + (unsigned)ln;
            unsigned h = nn / CDIM, c = nn - h * CDIM;
            float bias = biasp[nn];
            unsigned base = ((h * LDIM + l2b) * LDIM + l1u) * CP + c
                          + (unsigned)(half * 8) * (LDIM * CP);
#pragma unroll
            for (int e = 0; e < 8; ++e)
                dst[base + (unsigned)e * (LDIM * CP)] = (__bf16)(acc[e] + bias);
        }
    } else if (mmat == 2) {
        for (int it = 0; it < 9; ++it) {
            v8f acc = {};
            const __bf16* wcol = wbase + (unsigned)it * (16 * 128);
#pragma unroll
            for (int kc = 0; kc < 4; ++kc)
                acc = __builtin_amdgcn_wmma_f32_16x16x32_bf16(
                    false, afrag[kc], false, load_b_frag(wcol + kc * 32),
                    (short)0, acc, false, false);
            unsigned nn = (tb + (unsigned)it) * 16u + (unsigned)ln;
            unsigned h = nn / CDIM, c = nn - h * CDIM;
            float bias = bv[nn];
            // j-major v: rows of this tile are consecutive l2 -> one 16B store
            unsigned base = (h * 9216u + l1u * CDIM + c) * LDIM + l2b + half * 8;
            v8bf pack;
#pragma unroll
            for (int e = 0; e < 8; ++e) pack[e] = (__bf16)(acc[e] + bias);
            *reinterpret_cast<v8bf*>(va + base) = pack;
        }
    } else {
        for (int it = 0; it < 9; ++it) {
            v8f acc = {};
            const __bf16* wcol = wbase + (unsigned)it * (16 * 128);
#pragma unroll
            for (int kc = 0; kc < 4; ++kc)
                acc = __builtin_amdgcn_wmma_f32_16x16x32_bf16(
                    false, afrag[kc], false, load_b_frag(wcol + kc * 32),
                    (short)0, acc, false, false);
            unsigned nn = (tb + (unsigned)it) * 16u + (unsigned)ln;
            float bias = bg[nn];
            unsigned base = r0 * HC + nn + (unsigned)(half * 8) * HC;
#pragma unroll
            for (int e = 0; e < 8; ++e) {
                // sigmoid with fast v_rcp (result is bf16 anyway)
                float den = 1.0f + __expf(-(acc[e] + bias));
                gb[base + (unsigned)e * HC] = (__bf16)__builtin_amdgcn_rcpf(den);
            }
        }
    }

    // wave 0: the small bias-projection head (N = 12, padded to 16)
    if (wave == 0) {
        v8f acc = {};
        const __bf16* wcol = Wb16T + (unsigned)ln * 128u + half * 16;
#pragma unroll
        for (int kc = 0; kc < 4; ++kc)
            acc = __builtin_amdgcn_wmma_f32_16x16x32_bf16(
                false, afrag[kc], false, load_b_frag(wcol + kc * 32),
                (short)0, acc, false, false);
        if (ln < HHEADS) {
            float bias = bb2[ln];
            unsigned base = (r0 + half * 8) * HHEADS + (unsigned)ln;
#pragma unroll
            for (int e = 0; e < 8; ++e)
                projb[base + (unsigned)e * HHEADS] = acc[e] + bias;
        }
    }
}

// ---------- scores: att_h[l,k] = sum_{b,c} q*k, + bias, softmax over k ----------
__global__ __launch_bounds__(256) void attn_scores(
    const __bf16* qp, const __bf16* kp, const float* projb, __bf16* att)
{
    int lt = blockIdx.x, h = blockIdx.y;
    int l0 = lt * 16;
    __shared__ __attribute__((aligned(16))) float S[16][392];
    int t = threadIdx.x, wave = t >> 5, lane = t & 31, half = lane >> 4, ln = lane & 15;

    v8f acc0 = {}, acc1 = {}, acc2 = {};
    const __bf16* qptr = qp + (unsigned)((h * LDIM + l0 + ln)) * (LDIM * CP);
    const __bf16* kp0 = kp + (unsigned)((h * LDIM + (wave +  0) * 16 + ln)) * (LDIM * CP) + half * 16;
    const __bf16* kp1 = kp + (unsigned)((h * LDIM + (wave +  8) * 16 + ln)) * (LDIM * CP) + half * 16;
    const __bf16* kp2 = kp + (unsigned)((h * LDIM + (wave + 16) * 16 + ln)) * (LDIM * CP) + half * 16;

#pragma unroll 2
    for (int b = 0; b < LDIM; ++b) {
        v16bf a = load_a_frag(qptr, half);
        acc0 = __builtin_amdgcn_wmma_f32_16x16x32_bf16(false, a, false, load_b_frag(kp0),
                                                       (short)0, acc0, false, false);
        acc1 = __builtin_amdgcn_wmma_f32_16x16x32_bf16(false, a, false, load_b_frag(kp1),
                                                       (short)0, acc1, false, false);
        acc2 = __builtin_amdgcn_wmma_f32_16x16x32_bf16(false, a, false, load_b_frag(kp2),
                                                       (short)0, acc2, false, false);
        qptr += CP; kp0 += CP; kp1 += CP; kp2 += CP;
    }
    const float scale = 1.0f / 96.0f;  // 1/sqrt(C)/sqrt(L2) = 1/sqrt(9216)
#pragma unroll
    for (int e = 0; e < 8; ++e) {
        S[e + half * 8][(wave +  0) * 16 + ln] = acc0[e] * scale;
        S[e + half * 8][(wave +  8) * 16 + ln] = acc1[e] * scale;
        S[e + half * 8][(wave + 16) * 16 + ln] = acc2[e] * scale;
    }
    __syncthreads();

    // softmax over k for the 16 rows; 16 threads per row, 24 contiguous cols each
    int i = t >> 4, j = t & 15;
    int l = l0 + i;
    float x[24]; float mx = -1e30f;
#pragma unroll
    for (int c2 = 0; c2 < 24; ++c2) {
        unsigned k = (unsigned)(j * 24 + c2);
        float xx = S[i][k] + projb[(k * LDIM + l) * HHEADS + h];
        x[c2] = xx; mx = fmaxf(mx, xx);
    }
#pragma unroll
    for (int m = 8; m >= 1; m >>= 1) mx = fmaxf(mx, __shfl_xor(mx, m, 16));
    float sum = 0.f;
#pragma unroll
    for (int c2 = 0; c2 < 24; ++c2) { x[c2] = __expf(x[c2] - mx); sum += x[c2]; }
#pragma unroll
    for (int m = 8; m >= 1; m >>= 1) sum += __shfl_xor(sum, m, 16);
    float inv = __builtin_amdgcn_rcpf(sum);   // bf16 result: fast rcp is plenty
    v8bf o3[3];
#pragma unroll
    for (int c2 = 0; c2 < 24; ++c2) o3[c2 >> 3][c2 & 7] = (__bf16)(x[c2] * inv);
    v8bf* dst = reinterpret_cast<v8bf*>(
        att + (unsigned)((h * LDIM + l)) * LDIM + j * 24);
#pragma unroll
    for (int p = 0; p < 3; ++p) dst[p] = o3[p];
}

// ---------- AV: O_h[l,(b,c)] = att_h[l,:] @ V_h[:,(b,c)], then sigmoid-gate ----------
__global__ __launch_bounds__(256) void attn_av(
    const __bf16* att, const __bf16* va, const __bf16* gb, __bf16* go)
{
    int jg = blockIdx.x, lt = blockIdx.y, h = blockIdx.z;
    int t = threadIdx.x, wave = t >> 5, lane = t & 31, half = lane >> 4, ln = lane & 15;
    int l0 = lt * 16;
    unsigned j = (unsigned)((jg * 8 + wave) * 16 + ln);

    v8f acc = {};
    const __bf16* ab = att + (unsigned)((h * LDIM + l0 + ln)) * LDIM;
    const __bf16* vb = va + ((unsigned)h * 9216u + j) * LDIM + half * 16;
#pragma unroll
    for (int kc = 0; kc < 12; ++kc) {
        v16bf a = load_a_frag(ab + kc * 32, half);
        v16bf bf = load_b_frag(vb + kc * 32);
        acc = __builtin_amdgcn_wmma_f32_16x16x32_bf16(false, a, false, bf,
                                                      (short)0, acc, false, false);
    }
    unsigned bcol = j / CDIM, c = j - bcol * CDIM;
    unsigned base = (bcol * LDIM + (unsigned)(l0 + half * 8)) * HC + h * CDIM + c;
#pragma unroll
    for (int e = 0; e < 8; ++e) {
        unsigned ad = base + (unsigned)e * HC;
        go[ad] = (__bf16)((float)gb[ad] * acc[e]);
    }
}

// ---------- output projection: out[l2,l1,:] = (g*o)[r=(l1,l2),:] @ Wo + bo ----------
__global__ __launch_bounds__(256) void out_proj(
    const __bf16* go, const __bf16* woT, const float* bo, float* out)
{
    int t = threadIdx.x, wave = t >> 5, lane = t & 31, half = lane >> 4, ln = lane & 15;
    unsigned r0 = blockIdx.x * 16;
    unsigned l1u = r0 / LDIM;        // block-uniform
    unsigned l2b = r0 % LDIM;
    int n = wave * 16 + ln;

    v8f acc = {};
    const __bf16* ab = go + (r0 + (unsigned)ln) * HC;
    const __bf16* wb = woT + (unsigned)n * HC + half * 16;
#pragma unroll
    for (int kc = 0; kc < 9; ++kc) {
        v16bf a = load_a_frag(ab + kc * 32, half);
        v16bf bf = load_b_frag(wb + kc * 32);
        acc = __builtin_amdgcn_wmma_f32_16x16x32_bf16(false, a, false, bf,
                                                      (short)0, acc, false, false);
    }
    float bias = bo[n];
    unsigned base = ((l2b + (unsigned)(half * 8)) * LDIM + l1u) * DDIM + (unsigned)n;
#pragma unroll
    for (int e = 0; e < 8; ++e)
        out[base + (unsigned)e * (LDIM * DDIM)] = acc[e] + bias;
}

extern "C" void kernel_launch(void* const* d_in, const int* in_sizes, int n_in,
                              void* d_out, int out_size, void* d_ws, size_t ws_size,
                              hipStream_t stream) {
    const float* z    = (const float*)d_in[0];
    const float* ln_g = (const float*)d_in[1];
    const float* ln_b = (const float*)d_in[2];
    const float* Wq   = (const float*)d_in[3];
    const float* bq   = (const float*)d_in[4];
    const float* Wk   = (const float*)d_in[5];
    const float* bk   = (const float*)d_in[6];
    const float* Wv   = (const float*)d_in[7];
    const float* bv   = (const float*)d_in[8];
    const float* Wb   = (const float*)d_in[9];
    const float* bb   = (const float*)d_in[10];
    const float* Wg   = (const float*)d_in[11];
    const float* bg   = (const float*)d_in[12];
    const float* Wo   = (const float*)d_in[13];
    const float* bo   = (const float*)d_in[14];
    float* out = (float*)d_out;

    char* ws = (char*)d_ws;
    __bf16* WcatT = (__bf16*)(ws + OFF_WCAT);
    __bf16* Wb16T = (__bf16*)(ws + OFF_WB16);
    __bf16* WoT   = (__bf16*)(ws + OFF_WO16);
    __bf16* qp    = (__bf16*)(ws + OFF_QP);
    __bf16* kp    = (__bf16*)(ws + OFF_KP);
    __bf16* va    = (__bf16*)(ws + OFF_VA);
    __bf16* gb    = (__bf16*)(ws + OFF_GB);
    float*  pb    = (float*) (ws + OFF_PB);
    __bf16* att   = (__bf16*)(ws + OFF_ATT);
    __bf16* go    = (__bf16*)(ws + OFF_GO);

    int prep_elems = NQKVG * 128 + 16 * 128 + 128 * HC;
    prep_weights<<<(prep_elems + 255) / 256, 256, 0, stream>>>(
        Wq, Wk, Wv, Wg, Wb, Wo, WcatT, Wb16T, WoT);

    unsigned padrows = 12u * LDIM * LDIM;
    zero_pads<<<(padrows + 255) / 256, 256, 0, stream>>>(qp, kp);

    ln_qkvg<<<LL / 16, 256, 0, stream>>>(
        z, ln_g, ln_b, bq, bk, bv, bg, bb, WcatT, Wb16T, qp, kp, va, gb, pb);

    attn_scores<<<dim3(LDIM / 16, HHEADS), 256, 0, stream>>>(qp, kp, pb, att);

    attn_av<<<dim3(9216 / 128, LDIM / 16, HHEADS), 256, 0, stream>>>(att, va, gb, go);

    out_proj<<<LL / 16, 256, 0, stream>>>(go, WoT, bo, out);
}
